// MultiHeadAttention_63754494542487
// MI455X (gfx1250) — matrix operand order repaired and checked by reading, not compiled, not run on hardware
//
#include <hip/hip_runtime.h>
#include <hip/hip_bf16.h>
#include <math.h>

// ---------------------------------------------------------------------------
// MHA forward for MI455X (gfx1250, wave32, WMMA f32_16x16x32_f16)
// B=4, S=4096, D_MODEL=512, H=8, DH=64
// d_out = [ normed (4*4096*512 f32) | attn (4*8*4096*4096 f32) ]
// ---------------------------------------------------------------------------

typedef __attribute__((ext_vector_type(16))) _Float16 v16h;
typedef __attribute__((ext_vector_type(8)))  _Float16 v8h;
typedef __attribute__((ext_vector_type(8)))  float    v8f;

#define B_   4
#define S_   4096
#define DM   512
#define H_   8
#define DH   64

// workspace layout (bytes); total ~66 MB
#define WS_WT   0ull                        // 4 x (512*512) f16 transposed weights
#define WS_QH   (WS_WT  + 4ull*DM*DM*2ull)  // q f16 [BH][S][64]  (pre-scaled by 1/8)
#define WS_KH   (WS_QH  + 32ull*S_*DH*2ull) // k f16 [BH][S][64]
#define WS_VT   (WS_KH  + 32ull*S_*DH*2ull) // v f16 [BH][64][S]  (transposed)
#define WS_CTX  (WS_VT  + 32ull*S_*DH*2ull) // context f16 [B*S][512]

__device__ __forceinline__ int lane_id() { return (int)(threadIdx.x & 31u); }

// A fragment: 16x32 f16 tile at `tile` (row-major, leading dim ld in halves).
// lane L: row = L&15; elems 0..7 -> K = koff+0..7 ; elems 8..15 -> K = koff+16..23
// where koff = (L<16 ? 0 : 8).
__device__ __forceinline__ v16h load_A_h(const _Float16* tile, int ld) {
  int l = lane_id();
  const _Float16* p = tile + (size_t)(l & 15) * ld + ((l < 16) ? 0 : 8);
  v8h lo = *(const v8h*)(p);
  v8h hi = *(const v8h*)(p + 16);
  v16h a;
#pragma unroll
  for (int i = 0; i < 8; ++i) { a[i] = lo[i]; a[i + 8] = hi[i]; }
  return a;
}

// Same A fragment but sourced from f32 memory, converted to f16 on the fly.
__device__ __forceinline__ v16h load_A_f32(const float* tile, int ld) {
  int l = lane_id();
  const float* p = tile + (size_t)(l & 15) * ld + ((l < 16) ? 0 : 8);
  v16h a;
#pragma unroll
  for (int i = 0; i < 8; ++i) {
    a[i]     = (_Float16)p[i];
    a[i + 8] = (_Float16)p[16 + i];
  }
  return a;
}

// B fragment: 32x16 f16 tile; source matrix stored [N][K] row-major (ld in halves).
// lane L: col n = L&15; elems 0..7 -> K = koff+0..7 ; elems 8..15 -> K = koff+16..23,
// koff = (L<16 ? 0 : 8). Two 16B loads.
__device__ __forceinline__ v16h load_B_h(const _Float16* tile, int ld) {
  int l = lane_id();
  const _Float16* p = tile + (size_t)(l & 15) * ld + ((l < 16) ? 0 : 8);
  v8h lo = *(const v8h*)(p);
  v8h hi = *(const v8h*)(p + 16);
  v16h b;
#pragma unroll
  for (int i = 0; i < 8; ++i) { b[i] = lo[i]; b[i + 8] = hi[i]; }
  return b;
}

__device__ __forceinline__ v8f wmma_f16(v16h a, v16h b, v8f c) {
  // D = A(16x32) * B(32x16) + C, f32 accumulate
  return __builtin_amdgcn_wmma_f32_16x16x32_f16(false, a, false, b, (short)0, c,
                                                false, false);
}

// ---------------------------------------------------------------------------
// K0: transpose+convert one 512x512 f32 weight -> f16 [N][K]
// ---------------------------------------------------------------------------
__global__ void wt_kernel(const float* __restrict__ w, _Float16* __restrict__ wt) {
  int idx = blockIdx.x * blockDim.x + threadIdx.x;   // 262144 elements
  int k = idx & (DM - 1);
  int n = idx >> 9;
  wt[(size_t)n * DM + k] = (_Float16)w[(size_t)k * DM + n];
}

// ---------------------------------------------------------------------------
// K1: projection GEMM. X[16384x512] f32 @ Wt -> f16, head-major layouts.
// mode 0: q (scaled by 0.125) -> [bh][s][64]
// mode 1: k                  -> [bh][s][64]
// mode 2: v                  -> [bh][d][S] (transposed)
// one wave per 16x16 output tile; 4 waves / block.
// ---------------------------------------------------------------------------
__global__ void __launch_bounds__(128)
proj_kernel(const float* __restrict__ X, const _Float16* __restrict__ Wt,
            const float* __restrict__ bias, _Float16* __restrict__ dst, int mode) {
  int tile = blockIdx.x * 4 + (threadIdx.x >> 5);
  int m0 = (tile >> 5) * 16;          // 1024 row tiles
  int n0 = (tile & 31) * 16;          // 32 col tiles
  v8f c = {};
#pragma unroll 4
  for (int k0 = 0; k0 < DM; k0 += 32) {
    v16h a = load_A_f32(X + (size_t)m0 * DM + k0, DM);
    v16h b = load_B_h(Wt + (size_t)n0 * DM + k0, DM);
    c = wmma_f16(a, b, c);
  }
  int l = lane_id();
  int col = l & 15, rb = (l < 16) ? 0 : 8;
  int n = n0 + col;
  float bv = bias[n];
  int h = n >> 6, d = n & 63;
  float scale = (mode == 0) ? 0.125f : 1.0f;   // fold 1/sqrt(64) into q
#pragma unroll
  for (int e = 0; e < 8; ++e) {
    int m  = m0 + rb + e;
    int b_ = m >> 12, s = m & (S_ - 1);
    int bh = b_ * H_ + h;
    float v = (c[e] + bv) * scale;
    if (mode == 2) dst[((size_t)bh * DH + d) * S_ + s] = (_Float16)v;
    else           dst[((size_t)bh * S_ + s) * DH + d] = (_Float16)v;
  }
}

// ---------------------------------------------------------------------------
// K2: fused attention, block-cooperative.
// Block = 4 waves, all on the same (b,h); each wave owns 16 query rows
// (64 rows per block). K tiles (64 keys x 64 d, 8 KB f16) are staged into
// LDS ONCE per block per pass (V too in pass 3) -> 4x less L2 traffic than
// per-wave streaming. Pass1: row max. Pass2: row sum-exp. Pass3: normalized
// P -> attn (the mandatory 2.15 GB HBM write) and context += P @ V through
// an LDS P-transpose.
// ---------------------------------------------------------------------------
__global__ void __launch_bounds__(128)
attn_kernel(const _Float16* __restrict__ qh, const _Float16* __restrict__ kh,
            const _Float16* __restrict__ vt, float* __restrict__ attn,
            _Float16* __restrict__ ctx) {
  __shared__ __align__(32) _Float16 kbuf[64 * DH];      // 64 keys x 64 d
  __shared__ __align__(32) _Float16 vbuf[DH * 64];      // 64 d x 64 keys
  __shared__ __align__(32) _Float16 pbuf[4][16 * 32];   // per-wave P staging

  const int t    = threadIdx.x;
  const int wave = t >> 5;
  const int bh   = blockIdx.x >> 6;                     // 32 (b,h) pairs
  const int q0   = (blockIdx.x & 63) * 64 + wave * 16;  // 64 rows per block
  const int l    = lane_id();
  const int col  = l & 15, rb = (l < 16) ? 0 : 8;

  const _Float16* qbh = qh + (size_t)bh * S_ * DH;
  const _Float16* kbh = kh + (size_t)bh * S_ * DH;
  const _Float16* vbh = vt + (size_t)bh * DH * S_;

  // cooperative 8 KB tile copies: 512 x 16B chunks, 4 per thread
  auto stage_k = [&](int kk) {
#pragma unroll
    for (int i = 0; i < 4; ++i) {
      int flat = i * 128 + t;
      int row = flat >> 3, ch = (flat & 7) * 8;
      *(v8h*)&kbuf[row * DH + ch] =
          *(const v8h*)(kbh + (size_t)(kk + row) * DH + ch);
    }
  };
  auto stage_v = [&](int kk) {
#pragma unroll
    for (int i = 0; i < 4; ++i) {
      int flat = i * 128 + t;
      int row = flat >> 3, ch = (flat & 7) * 8;
      *(v8h*)&vbuf[row * 64 + ch] =
          *(const v8h*)(vbh + (size_t)row * S_ + kk + ch);
    }
  };

  v16h a0 = load_A_h(qbh + (size_t)q0 * DH + 0, DH);
  v16h a1 = load_A_h(qbh + (size_t)q0 * DH + 32, DH);

  // ---- pass 1: per-row max over all 4096 keys ----
  float mrow[8];
#pragma unroll
  for (int r = 0; r < 8; ++r) mrow[r] = -1e30f;
  for (int kk = 0; kk < S_; kk += 64) {
    __syncthreads();
    stage_k(kk);
    __builtin_prefetch(kbh + (size_t)(kk + 64) * DH, 0, 0);  // global_prefetch_b8
    __syncthreads();
#pragma unroll
    for (int kc = 0; kc < 4; ++kc) {
      v8f s = {};
      s = wmma_f16(a0, load_B_h(kbuf + kc * 16 * DH + 0, DH), s);
      s = wmma_f16(a1, load_B_h(kbuf + kc * 16 * DH + 32, DH), s);
#pragma unroll
      for (int r = 0; r < 8; ++r) mrow[r] = fmaxf(mrow[r], s[r]);
    }
  }
#pragma unroll
  for (int r = 0; r < 8; ++r)
    for (int off = 1; off < 16; off <<= 1)
      mrow[r] = fmaxf(mrow[r], __shfl_xor(mrow[r], off, 16));

  // ---- pass 2: per-row sum of exp(s - max) ----
  float lsum[8];
#pragma unroll
  for (int r = 0; r < 8; ++r) lsum[r] = 0.0f;
  for (int kk = 0; kk < S_; kk += 64) {
    __syncthreads();
    stage_k(kk);
    __builtin_prefetch(kbh + (size_t)(kk + 64) * DH, 0, 0);
    __syncthreads();
#pragma unroll
    for (int kc = 0; kc < 4; ++kc) {
      v8f s = {};
      s = wmma_f16(a0, load_B_h(kbuf + kc * 16 * DH + 0, DH), s);
      s = wmma_f16(a1, load_B_h(kbuf + kc * 16 * DH + 32, DH), s);
#pragma unroll
      for (int r = 0; r < 8; ++r) lsum[r] += __expf(s[r] - mrow[r]);
    }
  }
  float linv[8];
#pragma unroll
  for (int r = 0; r < 8; ++r) {
    for (int off = 1; off < 16; off <<= 1)
      lsum[r] += __shfl_xor(lsum[r], off, 16);
    linv[r] = 1.0f / lsum[r];
  }

  // ---- pass 3: write normalized attn, accumulate context = P @ V ----
  v8f acc[4] = {};
  float* attnrow = attn + ((size_t)bh * S_ + q0) * S_;
  _Float16* pw = pbuf[wave];
  for (int kk = 0; kk < S_; kk += 64) {
    __syncthreads();
    stage_k(kk);
    stage_v(kk);
    __builtin_prefetch(kbh + (size_t)(kk + 64) * DH, 0, 0);
    __builtin_prefetch(vbh + (size_t)(kk + 64), 0, 0);
    __syncthreads();
#pragma unroll
    for (int sub = 0; sub < 2; ++sub) {       // 32 keys per P fragment
      int kb32 = sub * 32;
#pragma unroll
      for (int half = 0; half < 2; ++half) {
        int k16 = kb32 + half * 16;
        v8f s = {};
        s = wmma_f16(a0, load_B_h(kbuf + (size_t)k16 * DH + 0, DH), s);
        s = wmma_f16(a1, load_B_h(kbuf + (size_t)k16 * DH + 32, DH), s);
#pragma unroll
        for (int r = 0; r < 8; ++r) {
          float p = __expf(s[r] - mrow[r]) * linv[r];
          attnrow[(size_t)(rb + r) * S_ + kk + k16 + col] = p;  // HBM write
          pw[(rb + r) * 32 + half * 16 + col] = (_Float16)p;    // LDS transpose
        }
      }
      // DS ops are in-order per wave; wait for P stores before A-frag reload
      asm volatile("s_wait_dscnt 0" ::: "memory");
      v16h ap = load_A_h(pw, 32);
#pragma unroll
      for (int nt = 0; nt < 4; ++nt)
        acc[nt] = wmma_f16(ap, load_B_h(vbuf + (size_t)(nt * 16) * 64 + kb32, 64),
                           acc[nt]);
    }
  }

  // context -> f16 [B*S][512] (heads interleaved for the out-proj GEMM)
  int b_ = bh >> 3, h = bh & 7;
  _Float16* crow = ctx + ((size_t)b_ * S_ + q0) * DM + h * DH;
#pragma unroll
  for (int nt = 0; nt < 4; ++nt)
#pragma unroll
    for (int e = 0; e < 8; ++e)
      crow[(size_t)(rb + e) * DM + nt * 16 + col] = (_Float16)acc[nt][e];
}

// ---------------------------------------------------------------------------
// K3: out = ctx @ WO + WO_b + Q  (f32 result into d_out normed region)
// ---------------------------------------------------------------------------
__global__ void __launch_bounds__(128)
outproj_kernel(const _Float16* __restrict__ ctx, const _Float16* __restrict__ Wt,
               const float* __restrict__ bias, const float* __restrict__ resid,
               float* __restrict__ out) {
  int tile = blockIdx.x * 4 + (threadIdx.x >> 5);
  int m0 = (tile >> 5) * 16;
  int n0 = (tile & 31) * 16;
  v8f c = {};
#pragma unroll 4
  for (int k0 = 0; k0 < DM; k0 += 32) {
    v16h a = load_A_h(ctx + (size_t)m0 * DM + k0, DM);
    v16h b = load_B_h(Wt + (size_t)n0 * DM + k0, DM);
    c = wmma_f16(a, b, c);
  }
  int l = lane_id();
  int col = l & 15, rb = (l < 16) ? 0 : 8;
  int n = n0 + col;
  float bv = bias[n];
#pragma unroll
  for (int e = 0; e < 8; ++e) {
    size_t m = (size_t)(m0 + rb + e);
    out[m * DM + n] = c[e] + bv + resid[m * DM + n];
  }
}

// ---------------------------------------------------------------------------
// K4: in-place LayerNorm per 512-wide row. One wave per row.
// ---------------------------------------------------------------------------
__global__ void __launch_bounds__(256)
ln_kernel(float* __restrict__ out, const float* __restrict__ g,
          const float* __restrict__ bb) {
  int row = blockIdx.x * 8 + (threadIdx.x >> 5);
  int l = lane_id();
  float* p = out + (size_t)row * DM;
  float x[16], sum = 0.f, sq = 0.f;
#pragma unroll
  for (int i = 0; i < 16; ++i) {
    x[i] = p[l + 32 * i];
    sum += x[i];
    sq  += x[i] * x[i];
  }
  for (int off = 1; off < 32; off <<= 1) {
    sum += __shfl_xor(sum, off, 32);
    sq  += __shfl_xor(sq, off, 32);
  }
  float mu  = sum * (1.0f / DM);
  float var = sq * (1.0f / DM) - mu * mu;   // population variance (ddof=0)
  float inv = rsqrtf(var + 1e-5f);
#pragma unroll
  for (int i = 0; i < 16; ++i) {
    int idx = l + 32 * i;
    p[idx] = (x[i] - mu) * inv * g[idx] + bb[idx];
  }
}

// ---------------------------------------------------------------------------
extern "C" void kernel_launch(void* const* d_in, const int* in_sizes, int n_in,
                              void* d_out, int out_size, void* d_ws, size_t ws_size,
                              hipStream_t stream) {
  const float* Q    = (const float*)d_in[0];
  const float* K    = (const float*)d_in[1];
  const float* V    = (const float*)d_in[2];
  const float* WQ_w = (const float*)d_in[3];
  const float* WQ_b = (const float*)d_in[4];
  const float* WK_w = (const float*)d_in[5];
  const float* WK_b = (const float*)d_in[6];
  const float* WV_w = (const float*)d_in[7];
  const float* WV_b = (const float*)d_in[8];
  const float* WO_w = (const float*)d_in[9];
  const float* WO_b = (const float*)d_in[10];
  const float* ln_g = (const float*)d_in[11];
  const float* ln_b = (const float*)d_in[12];

  char* ws = (char*)d_ws;
  _Float16* wt  = (_Float16*)(ws + WS_WT);
  _Float16* qh  = (_Float16*)(ws + WS_QH);
  _Float16* khp = (_Float16*)(ws + WS_KH);
  _Float16* vtr = (_Float16*)(ws + WS_VT);
  _Float16* ctx = (_Float16*)(ws + WS_CTX);

  float* outn = (float*)d_out;                        // [4,4096,512]
  float* attn = outn + (size_t)B_ * S_ * DM;          // [4,8,4096,4096]

  const size_t WTSZ = (size_t)DM * DM;                // halves per weight

  // weights -> f16 [N][K]
  wt_kernel<<<WTSZ / 256, 256, 0, stream>>>(WQ_w, wt + 0 * WTSZ);
  wt_kernel<<<WTSZ / 256, 256, 0, stream>>>(WK_w, wt + 1 * WTSZ);
  wt_kernel<<<WTSZ / 256, 256, 0, stream>>>(WV_w, wt + 2 * WTSZ);
  wt_kernel<<<WTSZ / 256, 256, 0, stream>>>(WO_w, wt + 3 * WTSZ);

  // q/k/v projections (32768 wave-tiles each, 4 waves per block)
  proj_kernel<<<8192, 128, 0, stream>>>(Q, wt + 0 * WTSZ, WQ_b, qh, 0);
  proj_kernel<<<8192, 128, 0, stream>>>(K, wt + 1 * WTSZ, WK_b, khp, 1);
  proj_kernel<<<8192, 128, 0, stream>>>(V, wt + 2 * WTSZ, WV_b, vtr, 2);

  // fused softmax-attention: 2048 blocks = 32 (b,h) x 64 blocks of 64 q-rows
  attn_kernel<<<2048, 128, 0, stream>>>(qh, khp, vtr, attn, ctx);

  // output projection + bias + residual
  outproj_kernel<<<8192, 128, 0, stream>>>(ctx, wt + 3 * WTSZ, WO_b, Q, outn);

  // layernorm in place
  ln_kernel<<<2048, 256, 0, stream>>>(outn, ln_g, ln_b);
}
